// SpatioTemporalAttentionBlock_90374701842823
// MI455X (gfx1250) — matrix operand order repaired and checked by reading, not compiled, not run on hardware
//
#include <hip/hip_runtime.h>
#include <hip/hip_bf16.h>

// ---------------------------------------------------------------------------
// SpatioTemporalAttentionBlock for MI455X (gfx1250, wave32, WMMA)
// B=2, C=512, T*H*W = N = 2048 tokens, 8 heads x 64 dim.
// All GEMM work goes through v_wmma_f32_16x16x32_f16 (f16 in, f32 acc).
// Attention is flash-style (online softmax), never materializing the 2048x2048
// score matrix.  All intermediates live in a ~30MB workspace (fits in L2).
// ---------------------------------------------------------------------------

typedef _Float16 half8  __attribute__((ext_vector_type(8)));
typedef _Float16 half16 __attribute__((ext_vector_type(16)));
typedef float    f32x8  __attribute__((ext_vector_type(8)));

#define BB    2
#define CC    512
#define NTOK  2048           // T*H*W
#define NH    8
#define HD    64
#define QSCALE 0.125f        // 1/sqrt(64)

// ---------------------------------------------------------------------------
// Kernel 1: LayerNorm over channels.  One token per lane: consecutive lanes
// read consecutive n along the [C, N] input -> fully coalesced.
// Writes xn (f32, residual path) and xh (f16, GEMM A-matrix) in [B*N, C].
// ---------------------------------------------------------------------------
__global__ void ln_kernel(const float* __restrict__ x,
                          const float* __restrict__ ln_w,
                          const float* __restrict__ ln_b,
                          float* __restrict__ xn,
                          _Float16* __restrict__ xh) {
  int tok = blockIdx.x * blockDim.x + threadIdx.x;     // 0 .. B*NTOK-1
  int b = tok >> 11;
  int n = tok & (NTOK - 1);
  const float* xp = x + (size_t)b * CC * NTOK + n;

  float s = 0.f, ss = 0.f;
  for (int c = 0; c < CC; ++c) {
    float v = xp[(size_t)c * NTOK];
    s += v; ss += v * v;
  }
  float mu  = s * (1.0f / CC);
  float var = ss * (1.0f / CC) - mu * mu;
  float inv = rsqrtf(var + 1e-5f);

  float*    xnp = xn + (size_t)tok * CC;
  _Float16* xhp = xh + (size_t)tok * CC;
  for (int c = 0; c < CC; ++c) {
    float v = xp[(size_t)c * NTOK];
    float y = (v - mu) * inv * ln_w[c] + ln_b[c];
    xnp[c] = y;
    xhp[c] = (_Float16)y;
  }
}

// ---------------------------------------------------------------------------
// Kernel 2: f32 -> f16 weight conversion.
// ---------------------------------------------------------------------------
__global__ void cvt_kernel(const float* __restrict__ a,
                           _Float16* __restrict__ o, int n) {
  int i = blockIdx.x * blockDim.x + threadIdx.x;
  if (i < n) o[i] = (_Float16)a[i];
}

// ---------------------------------------------------------------------------
// WMMA fragment loaders (contiguous k order, same for A and B):
//   A 16x32 : lane<16 -> row M=lane,    halves = K k0..k0+15 contiguous
//             lane>=16-> row M=lane-16, halves = K k0+16..k0+31 contiguous
//   B 32x16 : lane<16 -> col N=lane,    halves = K k0..k0+15 contiguous
//             lane>=16-> col N=lane-16, halves = K k0+16..k0+31 contiguous
// Both patterns are contiguous 16B / 32B loads from row-major storage.
// ---------------------------------------------------------------------------
__device__ __forceinline__ half16 load_a_frag(const _Float16* row, int k0, int hi) {
  half8 c0 = *(const half8*)(row + k0 + hi * 16);
  half8 c1 = *(const half8*)(row + k0 + 8 + hi * 16);
  half16 a;
#pragma unroll
  for (int i = 0; i < 8; ++i) { a[i] = c0[i]; a[i + 8] = c1[i]; }
  return a;
}

// ---------------------------------------------------------------------------
// Kernel 3: QKV GEMM.  out[row, o] = xh[row, :] . w_qkv[o, :] + b_qkv[o]
// M = B*N = 4096, K = 512, Nout = 1536.  One wave per 16x64 tile.
// Epilogue scatters into head-major q (pre-scaled), k, and transposed v.
// ---------------------------------------------------------------------------
__global__ void gemm_qkv_kernel(const _Float16* __restrict__ xh,
                                const _Float16* __restrict__ wh,   // [3C, C]
                                const float* __restrict__ bias,    // [3C]
                                _Float16* __restrict__ qh,         // [B,NH,N,HD]
                                _Float16* __restrict__ kh,         // [B,NH,N,HD]
                                _Float16* __restrict__ vt) {       // [B,NH,HD,N]
  const int lane = threadIdx.x & 31;
  const int lh = lane & 15, hi = lane >> 4;
  const int wid = blockIdx.x * (blockDim.x >> 5) + (threadIdx.x >> 5);
  const int numM = (BB * NTOK) / 16;          // 256
  const int mt = wid % numM;
  const int nt = wid / numM;                  // 0..23
  const int m0 = mt * 16, n0 = nt * 64;

  f32x8 acc[4] = {};
  const _Float16* arow = xh + (size_t)(m0 + lh) * CC;

  for (int k0 = 0; k0 < CC; k0 += 32) {
    half16 a = load_a_frag(arow, k0, hi);
#pragma unroll
    for (int sub = 0; sub < 4; ++sub) {
      const _Float16* wrow =
          wh + (size_t)(n0 + sub * 16 + lh) * CC + k0 + hi * 16;
      half16 bf = *(const half16*)wrow;
      acc[sub] = __builtin_amdgcn_wmma_f32_16x16x32_f16(
          false, a, false, bf, (short)0, acc[sub], false, false);
    }
  }

#pragma unroll
  for (int sub = 0; sub < 4; ++sub) {
    int o = n0 + sub * 16 + lh;
    int which = o >> 9;            // 0=q 1=k 2=v (uniform per 16-lane group)
    int rem = o & (CC - 1);
    int h = rem >> 6, d = rem & (HD - 1);
    float bo = bias[o];
#pragma unroll
    for (int i = 0; i < 8; ++i) {
      int row = m0 + i + 8 * hi;                    // C-frag row striping
      int b = row >> 11, n = row & (NTOK - 1);
      float v = acc[sub][i] + bo;
      size_t hb = (size_t)(b * NH + h);
      if (which == 0)      qh[(hb * NTOK + n) * HD + d] = (_Float16)(v * QSCALE);
      else if (which == 1) kh[(hb * NTOK + n) * HD + d] = (_Float16)v;
      else                 vt[(hb * HD + d) * NTOK + n] = (_Float16)v;
    }
  }
}

// ---------------------------------------------------------------------------
// Kernel 4: flash attention.  One wave per (b, h, 16-query tile); 32 KV keys
// per iteration.  S via 4 WMMAs, online softmax with shfl_xor reductions over
// the 16-lane halves (matching C-fragment row striping), P converted from
// C-layout to A-layout through a per-wave LDS tile, then P.V via 4 WMMAs.
// ---------------------------------------------------------------------------
__global__ void attn_kernel(const _Float16* __restrict__ qh,
                            const _Float16* __restrict__ kh,
                            const _Float16* __restrict__ vt,
                            _Float16* __restrict__ xattn) {   // [B*N, C] f16
  __shared__ __align__(16) _Float16 plds[4][16][32];   // 1KB per wave

  const int lane = threadIdx.x & 31;
  const int lh = lane & 15, hi = lane >> 4;
  const int warp = threadIdx.x >> 5;
  const int wid = blockIdx.x * 4 + warp;     // 0..2047
  const int qt = wid & 127;                  // 128 query tiles per (b,h)
  const int bh = wid >> 7;                   // 0..15

  const _Float16* qbase = qh + (size_t)bh * NTOK * HD;
  const _Float16* kbase = kh + (size_t)bh * NTOK * HD;
  const _Float16* vbase = vt + (size_t)bh * HD * NTOK;
  _Float16* pl = &plds[warp][0][0];

  // Preload the 16x64 query block as two A-fragments (K = 0..31, 32..63).
  half16 aq[2];
  {
    const _Float16* qrow = qbase + (size_t)(qt * 16 + lh) * HD;
#pragma unroll
    for (int t = 0; t < 2; ++t) aq[t] = load_a_frag(qrow, 32 * t, hi);
  }

  f32x8 oacc[4] = {};
  float m_run[8], l_run[8];
#pragma unroll
  for (int i = 0; i < 8; ++i) { m_run[i] = -1e30f; l_run[i] = 0.f; }

  for (int kv0 = 0; kv0 < NTOK; kv0 += 32) {
    // ---- S = q . k^T  (16 x 32 tile, f32) -------------------------------
    f32x8 s[2];
#pragma unroll
    for (int sub = 0; sub < 2; ++sub) {
      f32x8 a = {};
#pragma unroll
      for (int t = 0; t < 2; ++t) {
        const _Float16* krow =
            kbase + (size_t)(kv0 + sub * 16 + lh) * HD + 32 * t + hi * 16;
        half16 bk = *(const half16*)krow;
        a = __builtin_amdgcn_wmma_f32_16x16x32_f16(
            false, aq[t], false, bk, (short)0, a, false, false);
      }
      s[sub] = a;
    }

    // ---- online softmax -------------------------------------------------
#pragma unroll
    for (int i = 0; i < 8; ++i) {
      float t = fmaxf(s[0][i], s[1][i]);
      t = fmaxf(t, __shfl_xor(t, 1));
      t = fmaxf(t, __shfl_xor(t, 2));
      t = fmaxf(t, __shfl_xor(t, 4));
      t = fmaxf(t, __shfl_xor(t, 8));
      float m_new = fmaxf(m_run[i], t);
      float corr = __expf(m_run[i] - m_new);
      float p0 = __expf(s[0][i] - m_new);
      float p1 = __expf(s[1][i] - m_new);
      float rs = p0 + p1;
      rs += __shfl_xor(rs, 1);
      rs += __shfl_xor(rs, 2);
      rs += __shfl_xor(rs, 4);
      rs += __shfl_xor(rs, 8);
      l_run[i] = l_run[i] * corr + rs;
      m_run[i] = m_new;
#pragma unroll
      for (int c = 0; c < 4; ++c) oacc[c][i] *= corr;
      int row = i + 8 * hi;                       // C-frag row for this lane
      pl[row * 32 + lh]      = (_Float16)p0;
      pl[row * 32 + 16 + lh] = (_Float16)p1;
    }
    asm volatile("s_wait_dscnt 0" ::: "memory");  // P stores -> P loads

    // ---- reload P in A-fragment layout ----------------------------------
    half16 ap;
    {
      half8 c0 = *(const half8*)(pl + lh * 32 + hi * 16);
      half8 c1 = *(const half8*)(pl + lh * 32 + 8 + hi * 16);
#pragma unroll
      for (int i = 0; i < 8; ++i) { ap[i] = c0[i]; ap[i + 8] = c1[i]; }
    }

    // ---- O += P . V  (v stored transposed -> contiguous B-fragments) ----
#pragma unroll
    for (int c = 0; c < 4; ++c) {
      const _Float16* vrow =
          vbase + (size_t)(c * 16 + lh) * NTOK + kv0 + hi * 16;
      half16 bv = *(const half16*)vrow;
      oacc[c] = __builtin_amdgcn_wmma_f32_16x16x32_f16(
          false, ap, false, bv, (short)0, oacc[c], false, false);
    }
  }

  // ---- epilogue: O / l -> xattn[b*N+n, h*64 + d] ------------------------
  int b = bh >> 3, h = bh & (NH - 1);
#pragma unroll
  for (int i = 0; i < 8; ++i) {
    float inv = 1.0f / l_run[i];
    int n = qt * 16 + i + 8 * hi;
    _Float16* op = xattn + (size_t)(b * NTOK + n) * CC + h * HD;
#pragma unroll
    for (int c = 0; c < 4; ++c)
      op[c * 16 + lh] = (_Float16)(oacc[c][i] * inv);
  }
}

// ---------------------------------------------------------------------------
// Kernel 5: projection GEMM + bias + residual, writing [B,C,T,H,W] directly.
// M = 4096, K = 512, Nout = 512.  One wave per 16x64 tile.
// ---------------------------------------------------------------------------
__global__ void gemm_proj_kernel(const _Float16* __restrict__ xa,  // [B*N, C]
                                 const _Float16* __restrict__ wh,  // [C, C]
                                 const float* __restrict__ bias,   // [C]
                                 const float* __restrict__ xn,     // [B*N, C]
                                 float* __restrict__ out) {        // [B,C,N]
  const int lane = threadIdx.x & 31;
  const int lh = lane & 15, hi = lane >> 4;
  const int wid = blockIdx.x * (blockDim.x >> 5) + (threadIdx.x >> 5);
  const int numM = (BB * NTOK) / 16;          // 256
  const int mt = wid % numM;
  const int nt = wid / numM;                  // 0..7
  const int m0 = mt * 16, n0 = nt * 64;

  f32x8 acc[4] = {};
  const _Float16* arow = xa + (size_t)(m0 + lh) * CC;

  for (int k0 = 0; k0 < CC; k0 += 32) {
    half16 a = load_a_frag(arow, k0, hi);
#pragma unroll
    for (int sub = 0; sub < 4; ++sub) {
      const _Float16* wrow =
          wh + (size_t)(n0 + sub * 16 + lh) * CC + k0 + hi * 16;
      half16 bf = *(const half16*)wrow;
      acc[sub] = __builtin_amdgcn_wmma_f32_16x16x32_f16(
          false, a, false, bf, (short)0, acc[sub], false, false);
    }
  }

#pragma unroll
  for (int sub = 0; sub < 4; ++sub) {
    int c = n0 + sub * 16 + lh;
    float bo = bias[c];
#pragma unroll
    for (int i = 0; i < 8; ++i) {
      int row = m0 + i + 8 * hi;
      int b = row >> 11, n = row & (NTOK - 1);
      float v = acc[sub][i] + bo + xn[(size_t)row * CC + c];
      out[((size_t)b * CC + c) * NTOK + n] = v;   // [B,C,T,H,W] layout
    }
  }
}

// ---------------------------------------------------------------------------
// Host launcher.  Inputs (setup_inputs order):
//   0:x [B,C,T,H,W] f32   1:ln_w [C]   2:ln_b [C]
//   3:w_qkv [3C,C]        4:b_qkv [3C] 5:w_proj [C,C]   6:b_proj [C]
// Workspace layout (bytes, all 16B aligned), total ~30 MB:
// ---------------------------------------------------------------------------
extern "C" void kernel_launch(void* const* d_in, const int* in_sizes, int n_in,
                              void* d_out, int out_size, void* d_ws,
                              size_t ws_size, hipStream_t stream) {
  const float* x      = (const float*)d_in[0];
  const float* ln_w   = (const float*)d_in[1];
  const float* ln_b   = (const float*)d_in[2];
  const float* w_qkv  = (const float*)d_in[3];
  const float* b_qkv  = (const float*)d_in[4];
  const float* w_proj = (const float*)d_in[5];
  const float* b_proj = (const float*)d_in[6];

  char* ws = (char*)d_ws;
  float*    xn      = (float*)(ws + 0);                 //  8 MB  [B*N, C] f32
  _Float16* xh      = (_Float16*)(ws + 8388608);        //  4 MB  [B*N, C] f16
  _Float16* wqkv_h  = (_Float16*)(ws + 12582912);       // 1.5 MB
  _Float16* wproj_h = (_Float16*)(ws + 14155776);       // 0.5 MB
  _Float16* q_h     = (_Float16*)(ws + 14680064);       //  4 MB  [B,NH,N,HD]
  _Float16* k_h     = (_Float16*)(ws + 18874368);       //  4 MB  [B,NH,N,HD]
  _Float16* vt_h    = (_Float16*)(ws + 23068672);       //  4 MB  [B,NH,HD,N]
  _Float16* xa_h    = (_Float16*)(ws + 27262976);       //  4 MB  [B*N, C]

  // 1) LayerNorm (4096 tokens, one per lane)
  ln_kernel<<<(BB * NTOK) / 256, 256, 0, stream>>>(x, ln_w, ln_b, xn, xh);

  // 2) weight conversion
  cvt_kernel<<<(3 * CC * CC) / 256, 256, 0, stream>>>(w_qkv, wqkv_h, 3 * CC * CC);
  cvt_kernel<<<(CC * CC) / 256, 256, 0, stream>>>(w_proj, wproj_h, CC * CC);

  // 3) QKV GEMM: 256 M-tiles x 24 N-tiles = 6144 waves, 8 waves/block
  gemm_qkv_kernel<<<768, 256, 0, stream>>>(xh, wqkv_h, b_qkv, q_h, k_h, vt_h);

  // 4) flash attention: 2048 waves (B*NH*128 query tiles), 4 waves/block
  attn_kernel<<<512, 128, 0, stream>>>(q_h, k_h, vt_h, xa_h);

  // 5) proj GEMM + bias + residual + output transpose: 2048 waves
  gemm_proj_kernel<<<256, 256, 0, stream>>>(xa_h, wproj_h, b_proj, xn,
                                            (float*)d_out);
}
